// SelfAttentionHead_3298534883431
// MI455X (gfx1250) — matrix operand order repaired and checked by reading, not compiled, not run on hardware
//
#include <hip/hip_runtime.h>
#include <hip/hip_bf16.h>

// ---------------------------------------------------------------------------
// SelfAttentionHead for MI455X (gfx1250): bf16 WMMA flash attention with
// TDM (tensor_load_to_lds) + async-to-LDS staging.
// B=8, T=2048, C=1024, H=128.  out = softmax(mask(QK^T * C^-0.5)) V
// ---------------------------------------------------------------------------

#define BB 8
#define TT 2048
#define CC 1024
#define HH 128
#define NROWS (BB * TT)          // 16384 flattened (b,t) rows

typedef __attribute__((ext_vector_type(16))) __bf16 v16bf;
typedef __attribute__((ext_vector_type(8)))  float  v8f;
typedef __attribute__((ext_vector_type(4)))  unsigned v4u;
typedef __attribute__((ext_vector_type(8)))  unsigned v8u;

union Frag { unsigned u[8]; v16bf v; };

__device__ __forceinline__ unsigned short f2bf(float f) {
  union { float f; unsigned u; } x; x.f = f;
  unsigned u = x.u;
  return (unsigned short)((u + 0x7fffu + ((u >> 16) & 1u)) >> 16);  // RNE
}

__device__ __forceinline__ unsigned rl(unsigned x) {
  return (unsigned)__builtin_amdgcn_readfirstlane((int)x);
}

// Issue one TDM 2D tile load: 32 rows x 128 bf16, row stride 128 elements,
// global -> LDS row-major.  D# per CDNA5 ISA.
__device__ __forceinline__ void tdm_load_k_tile(unsigned lds_addr, const void* gptr) {
  unsigned long long ga = (unsigned long long)gptr;
  v4u g0;
  g0.x = rl(1u);                                    // count=1, user mode
  g0.y = rl(lds_addr);                              // lds_addr[31:0]
  g0.z = rl((unsigned)(ga & 0xffffffffu));          // global_addr[31:0]
  g0.w = rl((unsigned)((ga >> 32) & 0x01ffffffu) | 0x80000000u); // ga[56:32] | type=2
  v8u g1;
  g1.s0 = rl(0x00010000u);                          // wg_mask=0, data_size=1 (2B)
  g1.s1 = rl(128u << 16);                           // tensor_dim0 = 128
  g1.s2 = rl((unsigned)TT << 16);                   // tensor_dim1 = 2048
  g1.s3 = rl(128u << 16);                           // tile_dim0 = 128
  g1.s4 = rl(32u);                                  // tile_dim1 = 32 (tile_dim2=0)
  g1.s5 = rl(128u);                                 // tensor_dim0_stride = 128
  g1.s6 = rl(0u);
  g1.s7 = rl(0u);
  asm volatile("tensor_load_to_lds %0, %1" : : "s"(g0), "s"(g1) : "memory");
}

// ---------------------------------------------------------------------------
// Kernel 1: QKV projection.  Per block: 32 rows x (H=128) of Q,K,V.
// grid = NROWS/32 = 512, block = 128 (4 waves).
// Q is pre-scaled by C^-0.5 = 1/32 (exact power of two).
// ---------------------------------------------------------------------------
__global__ __launch_bounds__(128) void qkv_proj_kernel(
    const float* __restrict__ emb,
    const float* __restrict__ Wq, const float* __restrict__ Wk,
    const float* __restrict__ Wv,
    unsigned short* __restrict__ Qws, unsigned short* __restrict__ Kws,
    unsigned short* __restrict__ Vws)
{
  __shared__ unsigned short Ebf[32][32];       // [m][klocal]   2 KB
  __shared__ unsigned short Wt[3][HH][32];     // [mat][n][klocal] transposed, 24 KB

  const int tid  = threadIdx.x;
  const int wave = tid >> 5;
  const int lane = tid & 31;
  const int half = lane >> 4;
  const int l16  = lane & 15;
  const long rowbase = (long)blockIdx.x * 32;
  const int mt = wave & 1;                     // wave's M-tile

  v8f acc[12];
#pragma unroll
  for (int i = 0; i < 12; ++i) acc[i] = {};

  for (int c0 = 0; c0 < CC; c0 += 32) {
    // --- stage embeddings tile 32x32 f32 -> bf16 LDS ---
#pragma unroll
    for (int it = 0; it < 2; ++it) {
      int idx = tid + it * 128;
      int row = idx >> 3;
      int f4  = idx & 7;
      const float4 vv = *(const float4*)(emb + (rowbase + row) * CC + c0 + f4 * 4);
      Ebf[row][f4 * 4 + 0] = f2bf(vv.x);
      Ebf[row][f4 * 4 + 1] = f2bf(vv.y);
      Ebf[row][f4 * 4 + 2] = f2bf(vv.z);
      Ebf[row][f4 * 4 + 3] = f2bf(vv.w);
    }
    // --- stage Wq/Wk/Wv chunk rows [c0,c0+32) transposed into LDS ---
#pragma unroll
    for (int it = 0; it < 24; ++it) {
      int idx = tid + it * 128;
      int mat = idx >> 10;
      int rem = idx & 1023;
      int kl  = rem >> 5;
      int f4  = rem & 31;
      const float* wp = (mat == 0) ? Wq : ((mat == 1) ? Wk : Wv);
      const float4 vv = *(const float4*)(wp + (long)(c0 + kl) * HH + f4 * 4);
      int n = f4 * 4;
      Wt[mat][n + 0][kl] = f2bf(vv.x);
      Wt[mat][n + 1][kl] = f2bf(vv.y);
      Wt[mat][n + 2][kl] = f2bf(vv.z);
      Wt[mat][n + 3][kl] = f2bf(vv.w);
    }
    __syncthreads();

    Frag a;
#pragma unroll
    for (int j = 0; j < 8; ++j) {
      int kl = (j < 4) ? (half * 8 + 2 * j) : (16 + half * 8 + 2 * (j - 4));
      a.u[j] = *(const unsigned*)&Ebf[mt * 16 + l16][kl];
    }

#pragma unroll
    for (int i = 0; i < 12; ++i) {
      int rest = (wave >> 1) + 2 * i;
      int mat  = rest % 3;
      int nt   = rest / 3;
      Frag bf;
#pragma unroll
      for (int j = 0; j < 8; ++j) {
        int kl = (j < 4) ? (half * 8 + 2 * j) : (16 + half * 8 + 2 * (j - 4));
        bf.u[j] = *(const unsigned*)&Wt[mat][nt * 16 + l16][kl];
      }
      acc[i] = __builtin_amdgcn_wmma_f32_16x16x32_bf16(
          false, a.v, false, bf.v, (short)0, acc[i], false, false);
    }
    __syncthreads();
  }

#pragma unroll
  for (int i = 0; i < 12; ++i) {
    int rest = (wave >> 1) + 2 * i;
    int mat  = rest % 3;
    int nt   = rest / 3;
    unsigned short* op = (mat == 0) ? Qws : ((mat == 1) ? Kws : Vws);
    float sc = (mat == 0) ? 0.03125f : 1.0f;
#pragma unroll
    for (int r = 0; r < 8; ++r) {
      long row = rowbase + mt * 16 + r + 8 * half;
      op[row * HH + nt * 16 + l16] = f2bf(acc[i][r] * sc);
    }
  }
}

// ---------------------------------------------------------------------------
// Kernel 2: causal flash attention.  K tile staged by TDM, V tile staged by
// async-to-LDS then transposed in LDS.  grid = (T/64, B), block = 128.
// ---------------------------------------------------------------------------
__global__ __launch_bounds__(128) void flash_attn_kernel(
    const unsigned short* __restrict__ Qws, const unsigned short* __restrict__ Kws,
    const unsigned short* __restrict__ Vws, float* __restrict__ out)
{
  __shared__ unsigned short Ks[32][HH];        // key tile, row-major (TDM dst)   8 KB
  __shared__ unsigned short Vrow[32][HH];      // value tile, row-major (async)   8 KB
  __shared__ unsigned short Vs[HH][34];        // value tile, transposed        ~8.5 KB
  __shared__ unsigned short Ps[4][16][32];     // per-wave P scratch              4 KB

  const int tid  = threadIdx.x;
  const int wave = tid >> 5;
  const int lane = tid & 31;
  const int half = lane >> 4;
  const int l16  = lane & 15;
  const int b    = blockIdx.y;
  const int q0b  = blockIdx.x * 64;
  const int q0   = q0b + wave * 16;
  const long rbase = (long)b * TT;

  const unsigned ks_base   = (unsigned)(size_t)&Ks[0][0];    // LDS byte offsets
  const unsigned vrow_base = (unsigned)(size_t)&Vrow[0][0];

  // --- load Q fragments once (4 K-chunks of 32 over H=128) ---
  Frag aq[4];
  {
    long m = rbase + q0 + l16;
#pragma unroll
    for (int cc = 0; cc < 4; ++cc)
#pragma unroll
      for (int j = 0; j < 8; ++j) {
        int kl = (j < 4) ? (half * 8 + 2 * j) : (16 + half * 8 + 2 * (j - 4));
        aq[cc].u[j] = *(const unsigned*)(Qws + m * HH + cc * 32 + kl);
      }
  }

  v8f o[8];
#pragma unroll
  for (int n = 0; n < 8; ++n) o[n] = {};
  float mrow[8], lrow[8];
#pragma unroll
  for (int r = 0; r < 8; ++r) { mrow[r] = -1e30f; lrow[r] = 0.0f; }

  const int kend = q0b + 64;
  for (int kt = 0; kt < kend; kt += 32) {
    // --- K tile: one TDM descriptor, issued by wave 0 only ---
    if (wave == 0) {
      tdm_load_k_tile(ks_base, Kws + (rbase + kt) * HH);
      __builtin_amdgcn_s_wait_tensorcnt(0);
    }
    // --- V tile: async copy to LDS row-major (64 B per thread) ---
    {
      unsigned long long vbase = (unsigned long long)(Vws + (rbase + kt) * HH);
#pragma unroll
      for (int it = 0; it < 4; ++it) {
        unsigned byteoff = (unsigned)(tid + it * 128) * 16u;   // 0..8176
        unsigned ldsa = vrow_base + byteoff;
        asm volatile("global_load_async_to_lds_b128 %0, %1, %2 offset:0"
                     : : "v"(ldsa), "v"(byteoff), "s"(vbase) : "memory");
      }
      asm volatile("s_wait_asynccnt 0x0" : : : "memory");
    }
    if (kt + 32 < kend) {   // prefetch next tiles (global_prefetch_b8)
      __builtin_prefetch((const char*)(Kws + (rbase + kt + 32) * HH) + tid * 64, 0, 1);
      __builtin_prefetch((const char*)(Vws + (rbase + kt + 32) * HH) + tid * 64, 0, 1);
    }
    __syncthreads();

    // --- transpose V tile in LDS: Vrow[key][h] -> Vs[h][key] ---
#pragma unroll
    for (int it = 0; it < 16; ++it) {
      int idx = tid + it * 128;                // dword 0..2047
      int row = idx >> 6;
      int h2  = idx & 63;
      unsigned v = *(const unsigned*)&Vrow[row][h2 * 2];
      Vs[h2 * 2 + 0][row] = (unsigned short)(v & 0xffffu);
      Vs[h2 * 2 + 1][row] = (unsigned short)(v >> 16);
    }
    __syncthreads();

    // --- S = Q K^T  (two 16-key N-tiles, 4 K-chunks each) ---
    v8f s[2]; s[0] = {}; s[1] = {};
#pragma unroll
    for (int nt = 0; nt < 2; ++nt)
#pragma unroll
      for (int cc = 0; cc < 4; ++cc) {
        Frag bk;
#pragma unroll
        for (int j = 0; j < 8; ++j) {
          int kl = (j < 4) ? (half * 8 + 2 * j) : (16 + half * 8 + 2 * (j - 4));
          bk.u[j] = *(const unsigned*)&Ks[nt * 16 + l16][cc * 32 + kl];
        }
        s[nt] = __builtin_amdgcn_wmma_f32_16x16x32_bf16(
            false, aq[cc].v, false, bk.v, (short)0, s[nt], false, false);
      }

    // --- causal mask + online softmax (16-lane row groups) ---
#pragma unroll
    for (int r = 0; r < 8; ++r) {
      int q = q0 + r + 8 * half;
      float s0 = s[0][r];
      float s1 = s[1][r];
      if (kt + l16 > q)      s0 = -1e30f;
      if (kt + 16 + l16 > q) s1 = -1e30f;
      float t = fmaxf(s0, s1);
#pragma unroll
      for (int mk = 1; mk < 16; mk <<= 1) t = fmaxf(t, __shfl_xor(t, mk, 32));
      float mnew  = fmaxf(mrow[r], t);
      float p0    = exp2f((s0 - mnew) * 1.442695041f);
      float p1    = exp2f((s1 - mnew) * 1.442695041f);
      float alpha = exp2f((mrow[r] - mnew) * 1.442695041f);
      mrow[r] = mnew;
      float ps = p0 + p1;
#pragma unroll
      for (int mk = 1; mk < 16; mk <<= 1) ps += __shfl_xor(ps, mk, 32);
      lrow[r] = lrow[r] * alpha + ps;
#pragma unroll
      for (int n = 0; n < 8; ++n) o[n][r] *= alpha;
      Ps[wave][r + 8 * half][l16]      = f2bf(p0);
      Ps[wave][r + 8 * half][16 + l16] = f2bf(p1);
    }

    // --- re-shape P into an A fragment (16x32 bf16) via per-wave LDS ---
    Frag ap;
#pragma unroll
    for (int j = 0; j < 8; ++j) {
      int kl = (j < 4) ? (half * 8 + 2 * j) : (16 + half * 8 + 2 * (j - 4));
      ap.u[j] = *(const unsigned*)&Ps[wave][l16][kl];
    }

    // --- O += P V  (8 H-tiles of 16) ---
#pragma unroll
    for (int n = 0; n < 8; ++n) {
      Frag bv;
#pragma unroll
      for (int j = 0; j < 8; ++j) {
        int kl = (j < 4) ? (half * 8 + 2 * j) : (16 + half * 8 + 2 * (j - 4));
        bv.u[j] = *(const unsigned*)&Vs[n * 16 + l16][kl];
      }
      o[n] = __builtin_amdgcn_wmma_f32_16x16x32_bf16(
          false, ap.v, false, bv.v, (short)0, o[n], false, false);
    }
    __syncthreads();
  }

  // --- normalize and store fp32 output ---
#pragma unroll
  for (int n = 0; n < 8; ++n)
#pragma unroll
    for (int r = 0; r < 8; ++r) {
      long row = rbase + q0 + r + 8 * half;
      out[row * HH + n * 16 + l16] = o[n][r] / lrow[r];
    }
}

// ---------------------------------------------------------------------------
extern "C" void kernel_launch(void* const* d_in, const int* in_sizes, int n_in,
                              void* d_out, int out_size, void* d_ws, size_t ws_size,
                              hipStream_t stream) {
  const float* emb = (const float*)d_in[0];
  const float* Wq  = (const float*)d_in[1];
  const float* Wk  = (const float*)d_in[2];
  const float* Wv  = (const float*)d_in[3];
  float* out = (float*)d_out;

  // workspace: bf16 Q, K, V  (3 * 16384 * 128 * 2B = 12 MB)
  unsigned short* Qws = (unsigned short*)d_ws;
  unsigned short* Kws = Qws + (size_t)NROWS * HH;
  unsigned short* Vws = Kws + (size_t)NROWS * HH;

  qkv_proj_kernel<<<NROWS / 32, 128, 0, stream>>>(emb, Wq, Wk, Wv, Qws, Kws, Vws);
  flash_attn_kernel<<<dim3(TT / 64, BB), 128, 0, stream>>>(Qws, Kws, Vws, out);
}
